// DrugGCNModel_47502338293797
// MI455X (gfx1250) — matrix operand order repaired and checked by reading, not compiled, not run on hardware
//
#include <hip/hip_runtime.h>
#include <hip/hip_bf16.h>
#include <math.h>

typedef __attribute__((ext_vector_type(16))) __bf16 v16bf;
typedef __attribute__((ext_vector_type(8)))  float  v8f;

#define HID 256

__device__ __forceinline__ float lrelu01(float v){ return v > 0.f ? v : 0.01f*v; }

// ---------------------------------------------------------------------------
// WMMA GEMM: C[M,Nc] = act(A[M,K] @ B[K,Nc] (+ bias))
// act: 0 = none, 1 = +bias, 2 = +bias & leaky_relu(0.01)
// M % 16 == 0, Nc % 64 == 0. FAST requires K % 32 == 0 (=> 16B-aligned rows).
//
// Barrier/LDS-free; each wave owns a 16x64 tile (4 accumulators, shared A
// fragment). Explicit double-buffered software pipeline: while tile buffer 0
// is converted + fed to 4 WMMAs, tile buffer 1's 4 b128 A + 64 scalar B loads are in flight
// (and vice versa), so waits are partial (s_wait_loadcnt k) and covered by
// WMMA latency. Separate named buffers -> no register-rotation copies.
// Cross-wave A/B reuse is left to the 192MB L2.
// ---------------------------------------------------------------------------
template<bool FAST>
__global__ __launch_bounds__(128)
void gemm_bf16_kernel(const float* __restrict__ A, const float* __restrict__ B,
                      const float* __restrict__ bias, float* __restrict__ C,
                      int M, int K, int Nc, int act)
{
  const int lane = threadIdx.x & 31;
  const int wave = threadIdx.x >> 5;
  const int row0 = blockIdx.x * 16;
  const int n0   = blockIdx.y * 256 + wave * 64;
  if (n0 >= Nc) return;                       // no barriers -> early exit is safe
  const int m    = lane & 15;
  const int koff = (lane >> 4) << 3;
  const float* __restrict__ Arow = A + (size_t)(row0 + m) * K;
  v8f acc0 = {}, acc1 = {}, acc2 = {}, acc3 = {};
  const int ksteps = (K + 31) >> 5;

  float ta0[16], tb0[64];   // tile buffer 0 (raw fp32)
  float ta1[16], tb1[64];   // tile buffer 1 (raw fp32)

  auto loadT = [&](int k0, float (&ta)[16], float (&tb)[64]) {
    if (FAST) {
      const float4* p0 = reinterpret_cast<const float4*>(Arow + k0 + koff);
      const float4* p1 = reinterpret_cast<const float4*>(Arow + k0 + 16 + koff);
      float4 a0 = p0[0], a1 = p0[1], a2 = p1[0], a3 = p1[1];
      ta[0]=a0.x;  ta[1]=a0.y;  ta[2]=a0.z;  ta[3]=a0.w;
      ta[4]=a1.x;  ta[5]=a1.y;  ta[6]=a1.z;  ta[7]=a1.w;
      ta[8]=a2.x;  ta[9]=a2.y;  ta[10]=a2.z; ta[11]=a2.w;
      ta[12]=a3.x; ta[13]=a3.y; ta[14]=a3.z; ta[15]=a3.w;
      #pragma unroll
      for (int q = 0; q < 64; ++q) {
        const int s = q & 15;
        const int k = k0 + ((s < 8) ? (koff + s) : (16 + koff + s - 8));
        tb[q] = B[(size_t)k * Nc + n0 + 16*(q >> 4) + m];
      }
    } else {
      #pragma unroll
      for (int i = 0; i < 8; ++i) { int k = k0 + koff + i;      ta[i]   = (k < K) ? Arow[k] : 0.f; }
      #pragma unroll
      for (int i = 0; i < 8; ++i) { int k = k0 + 16 + koff + i; ta[8+i] = (k < K) ? Arow[k] : 0.f; }
      #pragma unroll
      for (int e = 0; e < 64; ++e) {
        const int s = e & 15;
        const int kb = k0 + ((s < 8) ? (koff + s) : (16 + koff + s - 8));
        tb[e] = (kb < K) ? B[(size_t)kb * Nc + n0 + 16*(e >> 4) + m] : 0.f;
      }
    }
  };

  auto mma = [&](float (&ta)[16], float (&tb)[64]) {
    v16bf av, bv0, bv1, bv2, bv3;
    #pragma unroll
    for (int i = 0; i < 16; ++i) av[i] = (__bf16)ta[i];
    #pragma unroll
    for (int e = 0; e < 16; ++e) {
      bv0[e] = (__bf16)tb[e];      bv1[e] = (__bf16)tb[16+e];
      bv2[e] = (__bf16)tb[32+e];   bv3[e] = (__bf16)tb[48+e];
    }
    acc0 = __builtin_amdgcn_wmma_f32_16x16x32_bf16(false, av, false, bv0, (short)0, acc0, false, false);
    acc1 = __builtin_amdgcn_wmma_f32_16x16x32_bf16(false, av, false, bv1, (short)0, acc1, false, false);
    acc2 = __builtin_amdgcn_wmma_f32_16x16x32_bf16(false, av, false, bv2, (short)0, acc2, false, false);
    acc3 = __builtin_amdgcn_wmma_f32_16x16x32_bf16(false, av, false, bv3, (short)0, acc3, false, false);
  };

  loadT(0, ta0, tb0);
  int ks = 0;
  for (; ks + 2 <= ksteps; ks += 2) {
    loadT((ks + 1) << 5, ta1, tb1);     // buffer 1 loads fly under buffer 0 math
    mma(ta0, tb0);
    if (ks + 2 < ksteps) loadT((ks + 2) << 5, ta0, tb0);  // and vice versa
    mma(ta1, tb1);
  }
  if (ks < ksteps) mma(ta0, tb0);       // odd-ksteps tail (tile already resident)

  // C/D layout: lane L, VGPR v -> row = row0 + v + 8*(L/16), col = n + (L%16)
  const int cn = lane & 15;
  const int mb = row0 + ((lane >> 4) << 3);
  #pragma unroll
  for (int j = 0; j < 4; ++j) {
    v8f acc = (j == 0) ? acc0 : (j == 1) ? acc1 : (j == 2) ? acc2 : acc3;
    const int col  = n0 + 16*j + cn;
    const float bc = (act >= 1) ? bias[col] : 0.f;
    #pragma unroll
    for (int v = 0; v < 8; ++v) {
      float val = acc[v] + bc;
      if (act == 2) val = lrelu01(val);
      C[(size_t)(mb + v) * Nc + col] = val;
    }
  }
}

// ---------------------------------------------------------------------------
__global__ __launch_bounds__(256)
void fill_kernel(float* p, float v, int n){
  int i = blockIdx.x*256 + threadIdx.x;
  if (i < n) p[i] = v;
}

// out[r] = dot(A[r,0:256], vec)
__global__ __launch_bounds__(256)
void rowdot_kernel(const float* __restrict__ A, const float* __restrict__ vec,
                   float* __restrict__ out, int R){
  int r = blockIdx.x*8 + (threadIdx.x>>5);
  if (r >= R) return;
  int lane = threadIdx.x & 31;
  const float* row = A + (size_t)r * HID;
  float p = 0.f;
  #pragma unroll
  for (int c = 0; c < 8; ++c) p += row[lane + 32*c] * vec[lane + 32*c];
  #pragma unroll
  for (int off = 16; off; off >>= 1) p += __shfl_xor(p, off, 32);
  if (lane == 0) out[r] = p;
}

// GATEConv logits: alpha[e] = lrelu( dot(lrelu(xWx[src] + edge_attr[e]@We), att_l) + xr[dst] )
__global__ __launch_bounds__(256)
void gate_alpha_kernel(const int* __restrict__ src, const int* __restrict__ dst,
                       const float* __restrict__ xWx, const float* __restrict__ eattr,
                       const float* __restrict__ We, const float* __restrict__ attl,
                       const float* __restrict__ xr, float* __restrict__ alpha, int E){
  int e = blockIdx.x*8 + (threadIdx.x>>5);
  if (e >= E) return;
  int lane = threadIdx.x & 31;
  int s = src[e], d = dst[e];
  float ea[10];
  #pragma unroll
  for (int k = 0; k < 10; ++k) ea[k] = eattr[(size_t)e*10 + k];
  const float* xrow = xWx + (size_t)s * HID;
  float p = 0.f;
  #pragma unroll
  for (int c = 0; c < 8; ++c) {
    int ch = lane + 32*c;
    float v = xrow[ch];
    #pragma unroll
    for (int k = 0; k < 10; ++k) v += ea[k] * We[k*HID + ch];
    p += lrelu01(v) * attl[ch];
  }
  #pragma unroll
  for (int off = 16; off; off >>= 1) p += __shfl_xor(p, off, 32);
  if (lane == 0) alpha[e] = lrelu01(p + xr[d]);
}

// GATConv logits: alpha[e] = lrelu(a_s[src[e]] + a_d[dst[e]])
__global__ __launch_bounds__(256)
void pair_alpha_kernel(const float* a_s, const float* a_d, const int* src,
                       const int* dst, float* alpha, int n){
  int i = blockIdx.x*256 + threadIdx.x;
  if (i < n) alpha[i] = lrelu01(a_s[src[i]] + a_d[dst[i]]);
}

// readout logits: alpha[i] = lrelu(a_s[i] + a_g[batch[i]])
__global__ __launch_bounds__(256)
void node_alpha_kernel(const float* a_s, const float* a_g, const int* batch,
                       float* alpha, int n){
  int i = blockIdx.x*256 + threadIdx.x;
  if (i < n) alpha[i] = lrelu01(a_s[i] + a_g[batch[i]]);
}

__device__ __forceinline__ void atomicMaxF(float* addr, float v){
  if (v >= 0.f) atomicMax((int*)addr, __float_as_int(v));
  else          atomicMin((unsigned int*)addr, (unsigned int)__float_as_int(v));
}

__global__ __launch_bounds__(256)
void segmax_kernel(const float* alpha, const int* idx, float* m, int n){
  int i = blockIdx.x*256 + threadIdx.x;
  if (i < n) atomicMaxF(&m[idx[i]], alpha[i]);
}

__global__ __launch_bounds__(256)
void segexp_kernel(float* alpha, const int* idx, const float* m, float* s, int n){
  int i = blockIdx.x*256 + threadIdx.x;
  if (i < n) {
    float a = expf(alpha[i] - m[idx[i]]);
    alpha[i] = a;
    atomicAdd(&s[idx[i]], a);
  }
}

__global__ __launch_bounds__(256)
void segnorm_kernel(float* alpha, const int* idx, const float* s, int n){
  int i = blockIdx.x*256 + threadIdx.x;
  if (i < n) alpha[i] = alpha[i] / (s[idx[i]] + 1e-16f);
}

// out[dst[e], :] += vals[src[e], :] * w[e]   (wave per edge)
__global__ __launch_bounds__(256)
void scatter_edges_kernel(const int* src, const int* dst, const float* vals,
                          const float* w, float* out, int E){
  int e = blockIdx.x*8 + (threadIdx.x>>5);
  if (e >= E) return;
  int lane = threadIdx.x & 31;
  int s = src[e], d = dst[e];
  float we = w[e];
  const float* vrow = vals + (size_t)s * HID;
  float* orow = out + (size_t)d * HID;
  #pragma unroll
  for (int c = 0; c < 8; ++c) {
    int ch = lane + 32*c;
    atomicAdd(&orow[ch], vrow[ch] * we);
  }
}

// out[idx[r], :] += vals[r, :] * (w ? w[r] : 1)
__global__ __launch_bounds__(256)
void scatter_rows_kernel(const int* idx, const float* vals, const float* w,
                         float* out, int R){
  int r = blockIdx.x*8 + (threadIdx.x>>5);
  if (r >= R) return;
  int lane = threadIdx.x & 31;
  int d = idx[r];
  float we = (w != nullptr) ? w[r] : 1.f;
  const float* vrow = vals + (size_t)r * HID;
  float* orow = out + (size_t)d * HID;
  #pragma unroll
  for (int c = 0; c < 8; ++c) {
    int ch = lane + 32*c;
    atomicAdd(&orow[ch], vrow[ch] * we);
  }
}

// mode 0: relu in-place; mode 1: elu(x + bias[col])
__global__ __launch_bounds__(256)
void biasact_kernel(float* d, const float* bias, int mode, int total){
  int i = blockIdx.x*256 + threadIdx.x;
  if (i >= total) return;
  float v = d[i];
  if (mode == 1) { v += bias[i & (HID-1)]; v = v > 0.f ? v : (expf(v) - 1.f); }
  else           { v = v > 0.f ? v : 0.f; }
  d[i] = v;
}

// fused GRUCell combine + relu; gi/gh already include their biases.
__global__ __launch_bounds__(256)
void gru_kernel(const float* __restrict__ gi, const float* __restrict__ gh,
                float* __restrict__ x, int R){
  int i = blockIdx.x*256 + threadIdx.x;
  if (i >= R*HID) return;
  int row = i >> 8, ch = i & 255;
  size_t b = (size_t)row*768 + ch;
  float r  = 1.f/(1.f + expf(-(gi[b]     + gh[b])));
  float z  = 1.f/(1.f + expf(-(gi[b+256] + gh[b+256])));
  float nn = tanhf(gi[b+512] + r*gh[b+512]);
  float o  = (1.f - z)*nn + z*x[i];
  x[i] = o > 0.f ? o : 0.f;
}

// ---------------------------------------------------------------------------
extern "C" void kernel_launch(void* const* d_in, const int* in_sizes, int n_in,
                              void* d_out, int out_size, void* d_ws, size_t ws_size,
                              hipStream_t stream) {
  (void)in_sizes; (void)n_in; (void)out_size; (void)ws_size;
  const int N = 131072, E = 524288, G = 4096;

  const float* x_in   = (const float*)d_in[0];
  const int*   ei     = (const int*)  d_in[1];
  const float* eattr  = (const float*)d_in[2];
  const int*   batch  = (const int*)  d_in[3];
  const float* lin1_w = (const float*)d_in[4];  const float* lin1_b = (const float*)d_in[5];
  const float* g_lin1_w=(const float*)d_in[6];  const float* g_lin2_w=(const float*)d_in[7];
  const float* g_att_l= (const float*)d_in[8];  const float* g_att_r= (const float*)d_in[9];
  const float* g_bias = (const float*)d_in[10];
  const float* gru0_wi= (const float*)d_in[11]; const float* gru0_wh= (const float*)d_in[12];
  const float* gru0_bi= (const float*)d_in[13]; const float* gru0_bh= (const float*)d_in[14];
  const float* conv_w = (const float*)d_in[15]; const float* conv_asrc=(const float*)d_in[16];
  const float* conv_adst=(const float*)d_in[17];const float* conv_b = (const float*)d_in[18];
  const float* agru_wi= (const float*)d_in[19]; const float* agru_wh= (const float*)d_in[20];
  const float* agru_bi= (const float*)d_in[21]; const float* agru_bh= (const float*)d_in[22];
  const float* mconv_w= (const float*)d_in[23]; const float* m_asrc = (const float*)d_in[24];
  const float* m_adst = (const float*)d_in[25]; const float* m_b    = (const float*)d_in[26];
  const float* mgru_wi= (const float*)d_in[27]; const float* mgru_wh= (const float*)d_in[28];
  const float* mgru_bi= (const float*)d_in[29]; const float* mgru_bh= (const float*)d_in[30];
  const float* lin2_w = (const float*)d_in[31]; const float* lin2_b = (const float*)d_in[32];

  const int* src = ei;        // edge_index[0]
  const int* dst = ei + E;    // edge_index[1]

  // workspace layout (fp32)
  float* ws    = (float*)d_ws;
  float* xb    = ws;                       // N x 256  current node features
  float* t0    = xb  + (size_t)N*256;      // N x 256
  float* t1    = t0  + (size_t)N*256;      // N x 256
  float* g0    = t1  + (size_t)N*256;      // N x 768  (gi)
  float* g1    = g0  + (size_t)N*768;      // N x 768  (gh)
  float* alpha = g1  + (size_t)N*768;      // E (also reused for N-length readout logits)
  float* segm  = alpha + E;                // N (segment max; G fits)
  float* segs  = segm + N;                 // N (segment sum)
  float* sc0   = segs + N;                 // N
  float* sc1   = sc0 + N;                  // N
  float* gout  = sc1 + N;                  // G x 256 graph state
  float* gtmp  = gout + (size_t)G*256;     // G x 256 (od / h)
  float* gg0   = gtmp + (size_t)G*256;     // G x 768
  float* gg1   = gg0  + (size_t)G*768;     // G x 768
  float* gscal = gg1  + (size_t)G*768;     // G

  auto blocks = [](int n){ return (n + 255) / 256; };
  auto gemm = [&](const float* A, const float* B, const float* bias, float* C,
                  int M, int K, int Nc, int act){
    dim3 grid(M/16, (Nc + 255) / 256);
    if ((K & 31) == 0)
      gemm_bf16_kernel<true ><<<grid, 128, 0, stream>>>(A, B, bias, C, M, K, Nc, act);
    else
      gemm_bf16_kernel<false><<<grid, 128, 0, stream>>>(A, B, bias, C, M, K, Nc, act);
  };
  auto segsoftmax = [&](float* a, const int* idx, int n, int nseg){
    fill_kernel<<<blocks(nseg),256,0,stream>>>(segm, -1e30f, nseg);
    fill_kernel<<<blocks(nseg),256,0,stream>>>(segs, 0.f, nseg);
    segmax_kernel <<<blocks(n),256,0,stream>>>(a, idx, segm, n);
    segexp_kernel <<<blocks(n),256,0,stream>>>(a, idx, segm, segs, n);
    segnorm_kernel<<<blocks(n),256,0,stream>>>(a, idx, segs, n);
  };

  // 1) x = lrelu(x_in @ lin1_w + lin1_b)            [N,39]@[39,256]
  gemm(x_in, lin1_w, lin1_b, xb, N, 39, 256, 2);

  // 2) GATEConv
  gemm(xb, g_lin1_w, nullptr, t0, N, 256, 256, 0);               // xWx (rows 0..255 of g_lin1_w)
  rowdot_kernel<<<N/8,256,0,stream>>>(xb, g_att_r, sc0, N);      // xr = x @ g_att_r
  gate_alpha_kernel<<<E/8,256,0,stream>>>(src, dst, t0, eattr,
                    g_lin1_w + 256*256, g_att_l, sc0, alpha, E);
  segsoftmax(alpha, dst, E, N);
  gemm(xb, g_lin2_w, nullptr, t1, N, 256, 256, 0);               // xg = x @ g_lin2_w
  fill_kernel<<<blocks(N*HID),256,0,stream>>>(t0, 0.f, N*HID);
  scatter_edges_kernel<<<E/8,256,0,stream>>>(src, dst, t1, alpha, t0, E);
  biasact_kernel<<<blocks(N*HID),256,0,stream>>>(t0, g_bias, 1, N*HID);   // h = elu(h + b)
  gemm(t0, gru0_wi, gru0_bi, g0, N, 256, 768, 1);                // gi = h @ wi + bi
  gemm(xb, gru0_wh, gru0_bh, g1, N, 256, 768, 1);                // gh = x @ wh + bh
  gru_kernel<<<blocks(N*HID),256,0,stream>>>(g0, g1, xb, N);     // x = relu(gru)

  // 3) GATConv layers
  for (int l = 0; l < 2; ++l) {
    gemm(xb, conv_w + (size_t)l*256*256, nullptr, t0, N, 256, 256, 0);   // xs
    rowdot_kernel<<<N/8,256,0,stream>>>(t0, conv_asrc + l*256, sc0, N);
    rowdot_kernel<<<N/8,256,0,stream>>>(t0, conv_adst + l*256, sc1, N);
    pair_alpha_kernel<<<blocks(E),256,0,stream>>>(sc0, sc1, src, dst, alpha, E);
    segsoftmax(alpha, dst, E, N);
    fill_kernel<<<blocks(N*HID),256,0,stream>>>(t1, 0.f, N*HID);
    scatter_edges_kernel<<<E/8,256,0,stream>>>(src, dst, t0, alpha, t1, E);
    biasact_kernel<<<blocks(N*HID),256,0,stream>>>(t1, conv_b + l*256, 1, N*HID);
    gemm(t1, agru_wi + (size_t)l*256*768, agru_bi + l*768, g0, N, 256, 768, 1);
    gemm(xb, agru_wh + (size_t)l*256*768, agru_bh + l*768, g1, N, 256, 768, 1);
    gru_kernel<<<blocks(N*HID),256,0,stream>>>(g0, g1, xb, N);
  }

  // 4) attentive readout
  fill_kernel<<<blocks(G*HID),256,0,stream>>>(gout, 0.f, G*HID);
  scatter_rows_kernel<<<N/8,256,0,stream>>>(batch, xb, nullptr, gout, N); // segment_sum
  biasact_kernel<<<blocks(G*HID),256,0,stream>>>(gout, nullptr, 0, G*HID); // relu
  gemm(xb, mconv_w, nullptr, t0, N, 256, 256, 0);                // xs (fixed across timesteps)
  rowdot_kernel<<<N/8,256,0,stream>>>(t0, m_asrc, sc0, N);       // a_src
  for (int t = 0; t < 2; ++t) {
    gemm(gout, mconv_w, nullptr, gtmp, G, 256, 256, 0);          // od
    rowdot_kernel<<<G/8,256,0,stream>>>(gtmp, m_adst, gscal, G);
    node_alpha_kernel<<<blocks(N),256,0,stream>>>(sc0, gscal, batch, alpha, N);
    segsoftmax(alpha, batch, N, G);
    fill_kernel<<<blocks(G*HID),256,0,stream>>>(gtmp, 0.f, G*HID);
    scatter_rows_kernel<<<N/8,256,0,stream>>>(batch, t0, alpha, gtmp, N);
    biasact_kernel<<<blocks(G*HID),256,0,stream>>>(gtmp, m_b, 1, G*HID);  // elu(h + m_b)
    gemm(gtmp, mgru_wi, mgru_bi, gg0, G, 256, 768, 1);
    gemm(gout, mgru_wh, mgru_bh, gg1, G, 256, 768, 1);
    gru_kernel<<<blocks(G*HID),256,0,stream>>>(gg0, gg1, gout, G);
  }

  // 5) final linear -> d_out [G,128]
  gemm(gout, lin2_w, lin2_b, (float*)d_out, G, 256, 128, 1);
}
